// PaiNN_21165598835328
// MI455X (gfx1250) — compile-verified, matrix operand order checked by reading, not run on hardware
//
#include <hip/hip_runtime.h>
#include <hip/hip_bf16.h>

// ---------------------------------------------------------------------------
// PaiNN forward for MI455X (gfx1250, wave32, WMMA).
//
// ~80 GFLOP total, dominated by the per-edge rbf@rbf_w GEMM
// (221184 x 64 x 384 per layer).  The [B,n,n,3F] edge tensor is never
// materialized in HBM; one workgroup per (b,i) computes the 48x384 filter
// block with WMMA into LDS (transposed, padded stride -> b64 stores,
// vectorized conflict-free reads) and reduces over senders j on-chip.
// All GEMMs use v_wmma_f32_16x16x32_bf16 with f32 accumulate; every weight
// (B-side) matrix is pre-packed into the ISA WMMA lane layout so a B
// fragment is one contiguous 32-byte load per lane.  upd_U/upd_V are packed
// side-by-side so one GEMM produces [Uv|Vv].  Unrolling is bounded so all
// kernels stay spill-free.
// ---------------------------------------------------------------------------

typedef __attribute__((ext_vector_type(16))) __bf16 bf16x16;
typedef __attribute__((ext_vector_type(8)))  __bf16 bf16x8;
typedef __attribute__((ext_vector_type(4)))  __bf16 bf16x4;
typedef __attribute__((ext_vector_type(8)))  float  f32x8;

#define BQ   96      // graphs
#define NA   48      // atoms per graph
#define NTOT (BQ*NA) // 4608 nodes
#define FD   128     // feature dim
#define F3   384     // 3*F
#define KR   64      // rbf dim
#define NL   6       // layers
#define CUTR 5.0f
#define WF_LD 52     // padded row stride of transposed Wf (26-bank stride)

__device__ __forceinline__ f32x8 wmma_bf16(bf16x16 a, bf16x16 b, f32x8 c) {
  return __builtin_amdgcn_wmma_f32_16x16x32_bf16(false, a, false, b, (short)0, c,
                                                 false, false);
}

// A fragment (16x32 of row-major [M][ld]); ISA 16-bit A layout:
// lanes 0-15: M=lane, K in {0..7} U {16..23}; lanes 16-31: M=lane-16, +8.
__device__ __forceinline__ bf16x16 load_frag_a(const __bf16* __restrict__ p,
                                               int ld, int lane) {
  int m  = lane & 15;
  int kb = (lane >> 4) * 8;
  const __bf16* row = p + (size_t)m * ld + kb;
  bf16x8 lo = *(const bf16x8*)(row);
  bf16x8 hi = *(const bf16x8*)(row + 16);
  return __builtin_shufflevector(lo, hi, 0, 1, 2, 3, 4, 5, 6, 7, 8, 9, 10, 11,
                                 12, 13, 14, 15);
}

// B fragment from pre-packed layout: tile base + lane*16 is contiguous.
__device__ __forceinline__ bf16x16 load_frag_bp(const __bf16* __restrict__ tile,
                                                int lane) {
  return *(const bf16x16*)(tile + lane * 16);
}

// ---------------------------------------------------------------------------
// Pack a row-major f32 [K][Nsrc] weight into bf16 WMMA-B fragment layout of a
// [K][Ndst] matrix at column offset colOff:
// P[((kt*NT + nt)*32 + lane)*16 + u], lane=(n&15)+16*((k&15)>>3),
// u=(k&7)+8*(k>>4 within 32-tile).  Layer slices of [L*K][N] stay contiguous
// (K % 32 == 0), so layer l lives at offset l*K*Ndst.
// ---------------------------------------------------------------------------
__global__ void pack_b_kernel(const float* __restrict__ B,
                              __bf16* __restrict__ P, int K, int Nsrc,
                              int Ndst, int colOff) {
  int i = blockIdx.x * 256 + threadIdx.x;
  if (i >= K * Nsrc) return;
  int k = i / Nsrc, n = i % Nsrc + colOff;
  int kt = k >> 5, kr = k & 31;
  int nt = n >> 4, nr = n & 15;
  int lane = nr + 16 * ((kr >> 3) & 1);
  int u    = (kr & 7) + 8 * (kr >> 4);
  int NT   = Ndst >> 4;
  P[(((size_t)kt * NT + nt) * 32 + lane) * 16 + u] = (__bf16)B[i];
}

// ---------------------------------------------------------------------------
// Tiled GEMM: D = silu?(A[M,KD]@B[KD,N] + bias); A bf16 row-major, B packed.
// Block = 128 threads (4 waves); block tile = 16 rows x 128 cols; each wave
// owns 2 col tiles (A-fragment feeds 2 WMMAs).  Outputs: f32 C (WF32) and/or
// bf16 shadow (WBF).  K-loop unroll bounded to 2 to avoid spills.
// ---------------------------------------------------------------------------
template <int KD, bool BIAS, bool SILU, bool WF32, bool WBF>
__global__ void gemm_kernel(const __bf16* __restrict__ A,
                            const __bf16* __restrict__ Bp,
                            const float* __restrict__ bias,
                            float* __restrict__ C, __bf16* __restrict__ Cbf,
                            int M, int N) {
  __shared__ __align__(16) __bf16 As[16 * KD];
  int m0   = blockIdx.y * 16;
  int wave = threadIdx.x >> 5;
  int lane = threadIdx.x & 31;
  int NT   = N >> 4;
  int nt   = blockIdx.x * 8 + wave * 2;

  // stage A tile (16 x KD bf16) with 16-byte vector copies
  {
    const uint4* ag  = (const uint4*)(A + (size_t)m0 * KD);
    uint4*       as4 = (uint4*)As;
#pragma unroll
    for (int i = threadIdx.x; i < 16 * KD / 8; i += 128) as4[i] = ag[i];
  }
  __syncthreads();

  f32x8 acc0 = {}, acc1 = {};
#pragma unroll 2
  for (int kt = 0; kt < KD; kt += 32) {
    bf16x16       af    = load_frag_a(As + kt, KD, lane);
    const __bf16* bbase = Bp + ((size_t)(kt >> 5) * NT + nt) * 512;
    bf16x16       b0    = load_frag_bp(bbase, lane);
    bf16x16       b1    = load_frag_bp(bbase + 512, lane);
    acc0 = wmma_bf16(af, b0, acc0);
    acc1 = wmma_bf16(af, b1, acc1);
  }

  int nr = lane & 15;
  int mo = (lane >> 4) * 8;
#pragma unroll
  for (int t = 0; t < 2; ++t) {
    int    nc  = (nt + t) * 16 + nr;
    float  bv  = BIAS ? bias[nc] : 0.0f;
    f32x8& acc = t ? acc1 : acc0;
#pragma unroll
    for (int u = 0; u < 8; ++u) {
      float val = acc[u] + bv;
      if (SILU) val = val / (1.0f + __expf(-val));
      size_t o = (size_t)(m0 + mo + u) * N + nc;
      if (WF32) C[o] = val;
      if (WBF) Cbf[o] = (__bf16)val;
    }
  }
}

// ---------------------------------------------------------------------------
// Geometry: per-graph pairwise d, cutoff envelope, unit dirs, bf16 RBF table.
// ---------------------------------------------------------------------------
__global__ void geom_kernel(const float* __restrict__ pos,
                            __bf16* __restrict__ rbf, float* __restrict__ wenv,
                            float* __restrict__ dirv) {
  int b = blockIdx.x;
  const float gamma = 1.0f / ((CUTR / KR) * (CUTR / KR) + 1e-9f);
  for (int p = threadIdx.x; p < NA * NA; p += blockDim.x) {
    int   i = p / NA, j = p % NA;
    int   gi = b * NA + i, gj = b * NA + j;
    float rx = pos[gi * 3 + 0] - pos[gj * 3 + 0];
    float ry = pos[gi * 3 + 1] - pos[gj * 3 + 1];
    float rz = pos[gi * 3 + 2] - pos[gj * 3 + 2];
    float d2   = rx * rx + ry * ry + rz * rz;
    bool  diag = (i == j);
    float d    = sqrtf(diag ? 1.0f : d2);
    float inv  = 1.0f / d;
    bool  msk  = (!diag) && (d < CUTR);
    float w = msk ? 0.5f * (__cosf(3.14159265358979f * d / CUTR) + 1.0f) : 0.0f;
    size_t eb = (size_t)gi * NA + j;
    wenv[eb] = w;
    dirv[eb * 3 + 0] = rx * inv;
    dirv[eb * 3 + 1] = ry * inv;
    dirv[eb * 3 + 2] = rz * inv;
    __bf16* rb = rbf + eb * KR;
    for (int k = 0; k < KR; ++k) {
      float c = CUTR * (float)k / (float)(KR - 1);
      float t = d - c;
      rb[k] = (__bf16)__expf(-gamma * t * t);
    }
  }
}

// ---------------------------------------------------------------------------
// Fused message kernel: block = one (b,i), 256 threads = 8 waves.
// Phase 1: stage rbf rows (48x64 bf16) + envelope + dirs in LDS.
// Phase 2: WMMA rbf@rbf_w (48x64x384) -> transposed WfT[col][row] in LDS
//          (stride WF_LD=52: lane's 8 accum values = 16 contiguous bytes ->
//          2x ds_store_b64; 26-bank thread stride -> conflict-free).
// Phase 3: j-reductions (ds / dvv / dvs); WfT rows contiguous in j ->
//          vectorized bf16x4 LDS loads; phi/v reads coalesced, L2-resident.
// ---------------------------------------------------------------------------
__global__ void message_kernel(const __bf16* __restrict__ rbf,
                               const float* __restrict__ wenv,
                               const float* __restrict__ dirv,
                               const float* __restrict__ phi,
                               const __bf16* __restrict__ rbfWp,
                               const float* __restrict__ rbfB,
                               const float* __restrict__ s_in,
                               const float* __restrict__ v_in,
                               float* __restrict__ s_out,
                               float* __restrict__ v_out) {
  __shared__ __align__(16) __bf16 rbfS[NA * KR];      // 6 KB
  __shared__ __align__(16) __bf16 WfT[F3 * WF_LD];    // 39 KB (transposed)
  __shared__ __align__(16) float  wS[NA];
  __shared__ __align__(16) float  dirS[NA * 3];

  int bi = blockIdx.x;  // b*NA + i
  int b  = bi / NA;

  {  // phase 1: vectorized staging
    const uint4* rg = (const uint4*)(rbf + (size_t)bi * NA * KR);
    uint4*       rs = (uint4*)rbfS;
#pragma unroll
    for (int i = threadIdx.x; i < NA * KR / 8; i += 256) rs[i] = rg[i];
    const uint4* wg = (const uint4*)(wenv + (size_t)bi * NA);
    uint4*       ws = (uint4*)wS;
    if (threadIdx.x < NA / 4) ws[threadIdx.x] = wg[threadIdx.x];
    const uint4* dg = (const uint4*)(dirv + (size_t)bi * NA * 3);
    uint4*       ds = (uint4*)dirS;
    if (threadIdx.x < NA * 3 / 4) ds[threadIdx.x] = dg[threadIdx.x];
  }
  __syncthreads();

  int wave = threadIdx.x >> 5;
  int lane = threadIdx.x & 31;
  int nr   = lane & 15;
  int mo   = (lane >> 4) * 8;
#pragma unroll 1
  for (int cc = 0; cc < 3; ++cc) {  // 8 waves x 3 = 24 col tiles
    int     cb = wave * 3 + cc;
    bf16x16 b0 = load_frag_bp(rbfWp + ((size_t)(0 * 24 + cb) * 32) * 16, lane);
    bf16x16 b1 = load_frag_bp(rbfWp + ((size_t)(1 * 24 + cb) * 32) * 16, lane);
    int   nc = cb * 16 + nr;
    float bv = rbfB[nc];
#pragma unroll 1
    for (int r = 0; r < 3; ++r) {
      f32x8   acc = {};
      bf16x16 a0  = load_frag_a(rbfS + (size_t)(r * 16) * KR + 0, KR, lane);
      bf16x16 a1  = load_frag_a(rbfS + (size_t)(r * 16) * KR + 32, KR, lane);
      acc = wmma_bf16(a0, b0, acc);
      acc = wmma_bf16(a1, b1, acc);
      union {
        __bf16 h[8];
        uint2  u2[2];
      } pk;
#pragma unroll
      for (int u = 0; u < 8; ++u) pk.h[u] = (__bf16)(acc[u] + bv);
      __bf16* dst = WfT + (size_t)nc * WF_LD + r * 16 + mo;
      *(uint2*)(dst)     = pk.u2[0];
      *(uint2*)(dst + 4) = pk.u2[1];
    }
  }
  __syncthreads();

  // phase 3: reductions over sender j (WfT rows contiguous in j)
  const float* phib = phi + (size_t)b * NA * F3;
  for (int idx = threadIdx.x; idx < 512; idx += 256) {
    if (idx < FD) {  // ds
      int           f    = idx;
      const __bf16* wrow = WfT + (size_t)f * WF_LD;
      float         acc  = 0.0f;
      for (int j0 = 0; j0 < NA; j0 += 4) {
        bf16x4 w4 = *(const bf16x4*)(wrow + j0);
#pragma unroll
        for (int u = 0; u < 4; ++u) {
          int j = j0 + u;
          acc += wS[j] * phib[(size_t)j * F3 + f] * (float)w4[u];
        }
      }
      size_t o = (size_t)bi * FD + f;
      s_out[o] = s_in[o] + acc;
    } else {  // dvv + dvs
      int           u2c   = idx - FD, c = u2c >> 7, f = u2c & 127;
      const __bf16* wrow1 = WfT + (size_t)(FD + f) * WF_LD;
      const __bf16* wrow2 = WfT + (size_t)(2 * FD + f) * WF_LD;
      float         acc   = 0.0f;
      for (int j0 = 0; j0 < NA; j0 += 4) {
        bf16x4 wa = *(const bf16x4*)(wrow1 + j0);
        bf16x4 wb = *(const bf16x4*)(wrow2 + j0);
#pragma unroll
        for (int u = 0; u < 4; ++u) {
          int   j  = j0 + u;
          float we = wS[j];
          acc += we * phib[(size_t)j * F3 + FD + f] * (float)wa[u] *
                 v_in[((size_t)(b * NA + j) * 3 + c) * FD + f];
          acc += we * phib[(size_t)j * F3 + 2 * FD + f] * (float)wb[u] *
                 dirS[j * 3 + c];
        }
      }
      size_t o = ((size_t)bi * 3 + c) * FD + f;
      v_out[o] = v_in[o] + acc;
    }
  }
}

// ---------------------------------------------------------------------------
// Small elementwise kernels.
// ---------------------------------------------------------------------------
__global__ void cast_kernel(const float* __restrict__ in,
                            __bf16* __restrict__ out, int n) {
  int i = blockIdx.x * 256 + threadIdx.x;
  if (i < n) out[i] = (__bf16)in[i];
}

__global__ void zero_kernel(float* __restrict__ p, int n) {
  int i = blockIdx.x * 256 + threadIdx.x;
  if (i < n) p[i] = 0.0f;
}

__global__ void embed_kernel(const int* __restrict__ node_atom,
                             const float* __restrict__ emb,
                             float* __restrict__ s, __bf16* __restrict__ sbf) {
  int i = blockIdx.x * 256 + threadIdx.x;
  if (i >= NTOT * FD) return;
  int   node = i / FD, f = i % FD;
  float v = emb[(size_t)node_atom[node] * FD + f];
  s[i]   = v;
  sbf[i] = (__bf16)v;
}

// Fused: catb[node][0:F] = bf16(s), catb[node][F:2F] = bf16(||Vv||_c)
// UvVv rows are [Uv(128) | Vv(128)] per (node,c).
__global__ void cat_kernel(const float* __restrict__ s,
                           const float* __restrict__ UvVv,
                           __bf16* __restrict__ cat) {
  int i = blockIdx.x * 256 + threadIdx.x;
  if (i >= NTOT * FD) return;
  int   node = i / FD, f = i % FD;
  float a = UvVv[((size_t)node * 3 + 0) * 2 * FD + FD + f];
  float b = UvVv[((size_t)node * 3 + 1) * 2 * FD + FD + f];
  float c = UvVv[((size_t)node * 3 + 2) * 2 * FD + FD + f];
  cat[(size_t)node * 2 * FD + f]      = (__bf16)s[i];
  cat[(size_t)node * 2 * FD + FD + f] = (__bf16)sqrtf(a * a + b * b + c * c + 1e-8f);
}

// Gated update; also emits the bf16 shadow of s (last writer of s per layer).
__global__ void apply_kernel(const float* __restrict__ a,
                             const float* __restrict__ UvVv,
                             float* __restrict__ s, float* __restrict__ v,
                             __bf16* __restrict__ sbf) {
  int i = blockIdx.x * 256 + threadIdx.x;
  if (i >= NTOT * FD) return;
  int   node = i / FD, f = i % FD;
  float avv = a[(size_t)node * F3 + f];
  float asv = a[(size_t)node * F3 + FD + f];
  float ass = a[(size_t)node * F3 + 2 * FD + f];
  float dot = 0.0f;
#pragma unroll
  for (int c = 0; c < 3; ++c) {
    size_t ro = ((size_t)node * 3 + c) * 2 * FD;
    float  u  = UvVv[ro + f];
    dot += u * UvVv[ro + FD + f];
    v[((size_t)node * 3 + c) * FD + f] += avv * u;
  }
  float sv = s[i] + asv * dot + ass;
  s[i]   = sv;
  sbf[i] = (__bf16)sv;
}

__global__ void out_reduce(const float* __restrict__ h,
                           const float* __restrict__ w2,
                           const float* __restrict__ b2,
                           float* __restrict__ out) {
  __shared__ float red[FD];
  int   b = blockIdx.x, f = threadIdx.x;
  float acc = 0.0f;
  for (int n = 0; n < NA; ++n) acc += h[((size_t)(b * NA + n)) * FD + f] * w2[f];
  red[f] = acc;
  __syncthreads();
  for (int s = FD / 2; s > 0; s >>= 1) {
    if (f < s) red[f] += red[f + s];
    __syncthreads();
  }
  if (f == 0) out[b] = red[0] + (float)NA * b2[0];
}

// ---------------------------------------------------------------------------
// Host launcher (stateless; all kernels on `stream`).
// ---------------------------------------------------------------------------
extern "C" void kernel_launch(void* const* d_in, const int* in_sizes, int n_in,
                              void* d_out, int out_size, void* d_ws,
                              size_t ws_size, hipStream_t stream) {
  const float* pos       = (const float*)d_in[1];
  const int*   node_atom = (const int*)d_in[3];
  const float* emb       = (const float*)d_in[4];
  const float* msg_w1 = (const float*)d_in[5];
  const float* msg_b1 = (const float*)d_in[6];
  const float* msg_w2 = (const float*)d_in[7];
  const float* msg_b2 = (const float*)d_in[8];
  const float* rbf_w  = (const float*)d_in[9];
  const float* rbf_b  = (const float*)d_in[10];
  const float* upd_U  = (const float*)d_in[11];
  const float* upd_V  = (const float*)d_in[12];
  const float* upd_w1 = (const float*)d_in[13];
  const float* upd_b1 = (const float*)d_in[14];
  const float* upd_w2 = (const float*)d_in[15];
  const float* upd_b2 = (const float*)d_in[16];
  const float* out_w1 = (const float*)d_in[17];
  const float* out_b1 = (const float*)d_in[18];
  const float* out_w2 = (const float*)d_in[19];
  const float* out_b2 = (const float*)d_in[20];

  char*  base = (char*)d_ws;
  size_t off  = 0;
  auto   alloc = [&](size_t bytes) -> char* {
    char* p = base + off;
    off     = (off + bytes + 255) & ~(size_t)255;
    return p;
  };

  // packed bf16 weights (WMMA-B fragment layout)
  __bf16* w_msg1 = (__bf16*)alloc((size_t)NL * FD * FD * 2);
  __bf16* w_msg2 = (__bf16*)alloc((size_t)NL * FD * F3 * 2);
  __bf16* w_rbf  = (__bf16*)alloc((size_t)NL * KR * F3 * 2);
  __bf16* w_UV   = (__bf16*)alloc((size_t)NL * FD * 2 * FD * 2);  // [Uv|Vv]
  __bf16* w_u1   = (__bf16*)alloc((size_t)NL * 2 * FD * FD * 2);
  __bf16* w_u2   = (__bf16*)alloc((size_t)NL * FD * F3 * 2);
  __bf16* w_o1   = (__bf16*)alloc((size_t)FD * FD * 2);

  float*  sA    = (float*)alloc((size_t)NTOT * FD * 4);
  float*  sB    = (float*)alloc((size_t)NTOT * FD * 4);
  float*  vA    = (float*)alloc((size_t)NTOT * 3 * FD * 4);
  float*  vB    = (float*)alloc((size_t)NTOT * 3 * FD * 4);
  __bf16* s_bf  = (__bf16*)alloc((size_t)NTOT * FD * 2);
  __bf16* v_bf  = (__bf16*)alloc((size_t)NTOT * 3 * FD * 2);
  float*  h     = (float*)alloc((size_t)NTOT * FD * 4);
  __bf16* h_bf  = (__bf16*)alloc((size_t)NTOT * FD * 2);
  float*  phi   = (float*)alloc((size_t)NTOT * F3 * 4);
  __bf16* rbfT  = (__bf16*)alloc((size_t)NTOT * NA * KR * 2);
  float*  wenv  = (float*)alloc((size_t)NTOT * NA * 4);
  float*  dirv  = (float*)alloc((size_t)NTOT * NA * 3 * 4);
  float*  UvVv  = (float*)alloc((size_t)NTOT * 3 * 2 * FD * 4);
  __bf16* catb  = (__bf16*)alloc((size_t)NTOT * 2 * FD * 2);
  __bf16* a1_bf = (__bf16*)alloc((size_t)NTOT * FD * 2);
  float*  amat  = (float*)alloc((size_t)NTOT * F3 * 4);

  auto pack = [&](const float* in, __bf16* out, int K, int Nsrc, int Ndst,
                  int colOff) {
    pack_b_kernel<<<(K * Nsrc + 255) / 256, 256, 0, stream>>>(in, out, K, Nsrc,
                                                              Ndst, colOff);
  };
  auto cast = [&](const float* in, __bf16* out, int n) {
    cast_kernel<<<(n + 255) / 256, 256, 0, stream>>>(in, out, n);
  };

  pack(msg_w1, w_msg1, NL * FD, FD, FD, 0);
  pack(msg_w2, w_msg2, NL * FD, F3, F3, 0);
  pack(rbf_w, w_rbf, NL * KR, F3, F3, 0);
  pack(upd_U, w_UV, NL * FD, FD, 2 * FD, 0);
  pack(upd_V, w_UV, NL * FD, FD, 2 * FD, FD);
  pack(upd_w1, w_u1, NL * 2 * FD, FD, FD, 0);
  pack(upd_w2, w_u2, NL * FD, F3, F3, 0);
  pack(out_w1, w_o1, FD, FD, FD, 0);

  geom_kernel<<<BQ, 256, 0, stream>>>(pos, rbfT, wenv, dirv);
  embed_kernel<<<(NTOT * FD + 255) / 256, 256, 0, stream>>>(node_atom, emb, sA,
                                                            s_bf);
  zero_kernel<<<(NTOT * 3 * FD + 255) / 256, 256, 0, stream>>>(vA, NTOT * 3 * FD);

  float* s_cur = sA;
  float* v_cur = vA;
  float* s_nxt = sB;
  float* v_nxt = vB;

  for (int l = 0; l < NL; ++l) {
    // phi = silu(s@w1+b1)@w2+b2   (s_bf = shadow of s_cur)
    gemm_kernel<FD, true, true, false, true>
        <<<dim3(FD / 128, NTOT / 16), 128, 0, stream>>>(
            s_bf, w_msg1 + (size_t)l * FD * FD, msg_b1 + (size_t)l * FD,
            nullptr, h_bf, NTOT, FD);
    gemm_kernel<FD, true, false, true, false>
        <<<dim3(F3 / 128, NTOT / 16), 128, 0, stream>>>(
            h_bf, w_msg2 + (size_t)l * FD * F3, msg_b2 + (size_t)l * F3, phi,
            nullptr, NTOT, F3);
    // fused edge filter + aggregation
    message_kernel<<<NTOT, 256, 0, stream>>>(
        rbfT, wenv, dirv, phi, w_rbf + (size_t)l * KR * F3,
        rbf_b + (size_t)l * F3, s_cur, v_cur, s_nxt, v_nxt);
    // update block: one GEMM produces [Uv|Vv]
    cast(v_nxt, v_bf, NTOT * 3 * FD);
    gemm_kernel<FD, false, false, true, false>
        <<<dim3(2 * FD / 128, NTOT * 3 / 16), 128, 0, stream>>>(
            v_bf, w_UV + (size_t)l * FD * 2 * FD, nullptr, UvVv, nullptr,
            NTOT * 3, 2 * FD);
    cat_kernel<<<(NTOT * FD + 255) / 256, 256, 0, stream>>>(s_nxt, UvVv, catb);
    gemm_kernel<2 * FD, true, true, false, true>
        <<<dim3(FD / 128, NTOT / 16), 128, 0, stream>>>(
            catb, w_u1 + (size_t)l * 2 * FD * FD, upd_b1 + (size_t)l * FD,
            nullptr, a1_bf, NTOT, FD);
    gemm_kernel<FD, true, false, true, false>
        <<<dim3(F3 / 128, NTOT / 16), 128, 0, stream>>>(
            a1_bf, w_u2 + (size_t)l * FD * F3, upd_b2 + (size_t)l * F3, amat,
            nullptr, NTOT, F3);
    apply_kernel<<<(NTOT * FD + 255) / 256, 256, 0, stream>>>(amat, UvVv,
                                                              s_nxt, v_nxt,
                                                              s_bf);
    // swap ping-pong buffers
    float* ts = s_cur; s_cur = s_nxt; s_nxt = ts;
    float* tv = v_cur; v_cur = v_nxt; v_nxt = tv;
  }

  // readout: out[b] = sum_n ( silu(s@w1+b1) @ w2 + b2 );  s_bf is current.
  gemm_kernel<FD, true, true, true, false>
      <<<dim3(FD / 128, NTOT / 16), 128, 0, stream>>>(s_bf, w_o1, out_b1, h,
                                                      nullptr, NTOT, FD);
  out_reduce<<<BQ, FD, 0, stream>>>(h, out_w2, out_b2, (float*)d_out);
}
